// SwitchTransformerEncoderLayer_90477781057908
// MI455X (gfx1250) — matrix operand order repaired and checked by reading, not compiled, not run on hardware
//
#include <hip/hip_runtime.h>
#include <hip/hip_bf16.h>

// ---------------- problem constants ----------------
#define S_   1024
#define B_   2
#define E_   1024
#define H_   16
#define HD_  64
#define DFF_ 4096
#define NE_  8
#define T_   (S_ * B_)     // 2048 tokens
#define PADR 128           // padding rows for expert-sorted buffers

// gfx1250 async global->LDS staging (set 0 to fall back to plain LDS stores)
#define USE_ASYNC_LDS 1

typedef __attribute__((ext_vector_type(16))) __bf16 v16bf;
typedef __attribute__((ext_vector_type(8)))  __bf16 v8bf;
typedef __attribute__((ext_vector_type(8)))  float  v8f;

// ---------------- scalar bf16 helpers (bit-level, RNE) ----------------
__device__ __forceinline__ unsigned short f2bf(float f) {
    unsigned int u = __float_as_uint(f);
    u += 0x7FFFu + ((u >> 16) & 1u);
    return (unsigned short)(u >> 16);
}
__device__ __forceinline__ float bf2f(unsigned short s) {
    return __uint_as_float(((unsigned int)s) << 16);
}

// ---------------- WMMA fragment loaders (CDNA5 ISA 7.12.2 layouts) ---------
// A matrix 16x32 bf16: lane<16 holds row (lane&15), K {k..k+7, k+16..k+23};
// lane>=16 holds K {k+8..k+15, k+24..k+31}.
__device__ __forceinline__ v16bf frag_a(const __bf16* p, int ld) {
    const int lane = threadIdx.x & 31;
    const int r  = lane & 15;
    const int k0 = (lane >> 4) * 8;
    const __bf16* row = p + (size_t)r * ld;
    v8bf lo = *(const v8bf*)(row + k0);
    v8bf hi = *(const v8bf*)(row + k0 + 16);
    v16bf out;
    #pragma unroll
    for (int i = 0; i < 8; ++i) { out[i] = lo[i]; out[i + 8] = hi[i]; }
    return out;
}
// Same pattern but from an LDS tile with fixed row pitch of 32 elements.
__device__ __forceinline__ v16bf frag_a_lds(const __bf16* p) {
    const int lane = threadIdx.x & 31;
    const int r  = lane & 15;
    const int k0 = (lane >> 4) * 8;
    const __bf16* row = p + r * 32;
    v8bf lo = *(const v8bf*)(row + k0);
    v8bf hi = *(const v8bf*)(row + k0 + 16);
    v16bf out;
    #pragma unroll
    for (int i = 0; i < 8; ++i) { out[i] = lo[i]; out[i + 8] = hi[i]; }
    return out;
}
// B matrix 32x16 bf16 (stored as W[n][k], k contiguous): lane<16 = col n,
// K {k..k+7, k+16..k+23}; lane>=16 = col n, K {k+8..k+15, k+24..k+31}.
__device__ __forceinline__ v16bf frag_b(const __bf16* p, int ld) {
    const int lane = threadIdx.x & 31;
    const int n  = lane & 15;
    const int k0 = (lane >> 4) * 8;
    const __bf16* col = p + (size_t)n * ld;
    v8bf lo = *(const v8bf*)(col + k0);
    v8bf hi = *(const v8bf*)(col + k0 + 16);
    v16bf out;
    #pragma unroll
    for (int i = 0; i < 8; ++i) { out[i] = lo[i]; out[i + 8] = hi[i]; }
    return out;
}

__device__ __forceinline__ void zero_acc(v8f acc[4][4]) {
    #pragma unroll
    for (int i = 0; i < 4; ++i)
        #pragma unroll
        for (int j = 0; j < 4; ++j)
            acc[i][j] = (v8f){0.f, 0.f, 0.f, 0.f, 0.f, 0.f, 0.f, 0.f};
}

// ---------------- direct-from-global 64x64 wave tile ----------------
__device__ __forceinline__ void wmma_block(const __bf16* Aw, const __bf16* Bw,
                                           int lda, int ldb, int K,
                                           v8f acc[4][4]) {
    #pragma unroll 1
    for (int kk = 0; kk < K; kk += 32) {
        v16bf a[4], b[4];
        #pragma unroll
        for (int i = 0; i < 4; ++i) a[i] = frag_a(Aw + (size_t)(i * 16) * lda + kk, lda);
        #pragma unroll
        for (int j = 0; j < 4; ++j) b[j] = frag_b(Bw + (size_t)(j * 16) * ldb + kk, ldb);
        #pragma unroll
        for (int i = 0; i < 4; ++i)
            #pragma unroll
            for (int j = 0; j < 4; ++j)
                acc[i][j] = __builtin_amdgcn_wmma_f32_16x16x32_bf16(
                    false, a[i], false, b[j], (short)0, acc[i][j], false, false);
    }
}

// ---------------- async-LDS staged variant (A tile shared per block) -------
// Stage a 128x32 bf16 A tile (8KB) into LDS: 512 16-byte chunks / 256 threads.
__device__ __forceinline__ void stage_a(const __bf16* Ablk, int lda, int kk,
                                        __bf16* lds) {
    const int tid = threadIdx.x;
    #pragma unroll
    for (int c = tid; c < 512; c += 256) {
        const int row = c >> 2;
        const int seg = c & 3;
        const __bf16* g = Ablk + (size_t)row * lda + kk + seg * 8;
        __bf16* l = lds + row * 32 + seg * 8;
#if USE_ASYNC_LDS
        // GLOBAL_LOAD_ASYNC_TO_LDS_B128: VDST = LDS byte address (low 32 bits
        // of generic LDS pointer per flat-aperture rules), VADDR = 64b global.
        asm volatile("global_load_async_to_lds_b128 %0, %1, off"
                     :: "v"((unsigned int)(unsigned long long)(uintptr_t)l),
                        "v"((unsigned long long)(uintptr_t)g)
                     : "memory");
#else
        *(v8bf*)l = *(const v8bf*)g;
#endif
    }
}
__device__ __forceinline__ void wait_async_lds() {
#if USE_ASYNC_LDS
#if __has_builtin(__builtin_amdgcn_s_wait_asynccnt)
    __builtin_amdgcn_s_wait_asynccnt(0);
#else
    asm volatile("s_wait_asynccnt 0" ::: "memory");
#endif
#endif
}

// A staged through double-buffered LDS; B streamed from global.
// Ablk = block's A base (128 rows), arow = this wave's row offset (wm*64).
__device__ __forceinline__ void wmma_block_async(const __bf16* Ablk, int arow,
                                                 const __bf16* Bw,
                                                 int lda, int ldb, int K,
                                                 __bf16* lds, v8f acc[4][4]) {
    stage_a(Ablk, lda, 0, lds);
    int buf = 0;
    #pragma unroll 1
    for (int kk = 0; kk < K; kk += 32) {
        wait_async_lds();
        __syncthreads();                         // staged tile visible to WG
        if (kk + 32 < K)
            stage_a(Ablk, lda, kk + 32, lds + (buf ^ 1) * (128 * 32));
        const __bf16* lbase = lds + buf * (128 * 32) + arow * 32;
        v16bf a[4], b[4];
        #pragma unroll
        for (int i = 0; i < 4; ++i) a[i] = frag_a_lds(lbase + i * 16 * 32);
        #pragma unroll
        for (int j = 0; j < 4; ++j) b[j] = frag_b(Bw + (size_t)(j * 16) * ldb + kk, ldb);
        #pragma unroll
        for (int i = 0; i < 4; ++i)
            #pragma unroll
            for (int j = 0; j < 4; ++j)
                acc[i][j] = __builtin_amdgcn_wmma_f32_16x16x32_bf16(
                    false, a[i], false, b[j], (short)0, acc[i][j], false, false);
        buf ^= 1;
    }
}

// ---------------- block reductions (wave32) ----------------
__device__ __forceinline__ float bredux(float v, float* sm, bool is_max) {
    #pragma unroll
    for (int m = 16; m; m >>= 1) {
        float o = __shfl_xor(v, m, 32);
        v = is_max ? fmaxf(v, o) : v + o;
    }
    const int wid = threadIdx.x >> 5;
    __syncthreads();
    if ((threadIdx.x & 31) == 0) sm[wid] = v;
    __syncthreads();
    if (threadIdx.x < 32) {
        const int nw = blockDim.x >> 5;
        float r = (threadIdx.x < (unsigned)nw) ? sm[threadIdx.x]
                                               : (is_max ? -3.0e38f : 0.0f);
        #pragma unroll
        for (int m = 16; m; m >>= 1) {
            float o = __shfl_xor(r, m, 32);
            r = is_max ? fmaxf(r, o) : r + o;
        }
        if (threadIdx.x == 0) sm[0] = r;
    }
    __syncthreads();
    return sm[0];
}

__device__ __forceinline__ float gelu_exact(float x) {
    return 0.5f * x * (1.0f + erff(x * 0.70710678118654752f));
}

// ===================== kernels =====================

__global__ __launch_bounds__(256) void k_cvt(const float* __restrict__ s,
                                             unsigned short* __restrict__ d, int n) {
    for (int i = blockIdx.x * blockDim.x + threadIdx.x; i < n;
         i += gridDim.x * blockDim.x)
        d[i] = f2bf(s[i]);
}

// QKV: [T,3E] = Xbf[T,E] @ Wbf[3E,E]^T + b ; scatter into q/k (B,H,S,HD) and
// v transposed (B,H,HD,S). q pre-scaled by 1/sqrt(HD).
__global__ __launch_bounds__(256) void k_qkv(const __bf16* __restrict__ X,
                                             const __bf16* __restrict__ W,
                                             const float* __restrict__ bias,
                                             unsigned short* __restrict__ q,
                                             unsigned short* __restrict__ kx,
                                             unsigned short* __restrict__ vt) {
    __shared__ __bf16 sA[2 * 128 * 32];
    const int wid = threadIdx.x >> 5, wm = wid >> 2, wn = wid & 3;
    const int mB = blockIdx.x * 128;
    const int m0 = mB + wm * 64;
    const int n0 = blockIdx.y * 256 + wn * 64;
    v8f acc[4][4]; zero_acc(acc);
    wmma_block_async(X + (size_t)mB * E_, wm * 64, W + (size_t)n0 * E_,
                     E_, E_, E_, sA, acc);
    const int lane = threadIdx.x & 31, col = lane & 15, rb = (lane >> 4) * 8;
    #pragma unroll
    for (int i = 0; i < 4; ++i)
        #pragma unroll
        for (int j = 0; j < 4; ++j)
            #pragma unroll
            for (int v = 0; v < 8; ++v) {
                const int t = m0 + i * 16 + rb + v;
                const int n = n0 + j * 16 + col;
                const float val = acc[i][j][v] + bias[n];
                const int sec = n >> 10, rem = n & 1023;
                const int h = rem >> 6, d = rem & 63;
                const int s = t >> 1, b = t & 1;   // t = s*B + b, B=2
                const size_t bh = (size_t)(b * H_ + h);
                if (sec == 0)      q[(bh * S_ + s) * HD_ + d] = f2bf(val * 0.125f);
                else if (sec == 1) kx[(bh * S_ + s) * HD_ + d] = f2bf(val);
                else               vt[(bh * HD_ + d) * S_ + s] = f2bf(val);
            }
}

// scores[z][s][t] = q[z][s,:] . k[z][t,:]   (K = 64), stored bf16
__global__ __launch_bounds__(256) void k_scores(const __bf16* __restrict__ q,
                                                const __bf16* __restrict__ kx,
                                                unsigned short* __restrict__ sc) {
    const int z = blockIdx.z;
    const int wid = threadIdx.x >> 5, wm = wid >> 2, wn = wid & 3;
    const int m0 = blockIdx.x * 128 + wm * 64;
    const int n0 = blockIdx.y * 256 + wn * 64;
    v8f acc[4][4]; zero_acc(acc);
    wmma_block(q + ((size_t)z * S_ + m0) * HD_,
               kx + ((size_t)z * S_ + n0) * HD_, HD_, HD_, HD_, acc);
    const int lane = threadIdx.x & 31, col = lane & 15, rb = (lane >> 4) * 8;
    #pragma unroll
    for (int i = 0; i < 4; ++i)
        #pragma unroll
        for (int j = 0; j < 4; ++j)
            #pragma unroll
            for (int v = 0; v < 8; ++v) {
                const int s = m0 + i * 16 + rb + v;
                const int t = n0 + j * 16 + col;
                sc[((size_t)z * S_ + s) * S_ + t] = f2bf(acc[i][j][v]);
            }
}

// row softmax over S_ elements, in-place bf16
__global__ __launch_bounds__(256) void k_softmax(unsigned short* __restrict__ sc) {
    __shared__ float sm[16];
    const size_t base = (size_t)blockIdx.x * S_;
    float v[4];
    float mx = -3.0e38f;
    #pragma unroll
    for (int i = 0; i < 4; ++i) {
        v[i] = bf2f(sc[base + threadIdx.x + i * 256]);
        mx = fmaxf(mx, v[i]);
    }
    mx = bredux(mx, sm, true);
    float sum = 0.f;
    #pragma unroll
    for (int i = 0; i < 4; ++i) { v[i] = __expf(v[i] - mx); sum += v[i]; }
    sum = bredux(sum, sm, false);
    const float inv = 1.0f / sum;
    #pragma unroll
    for (int i = 0; i < 4; ++i)
        sc[base + threadIdx.x + i * 256] = f2bf(v[i] * inv);
}

// O[z][s][d] = probs[z][s,:] @ v[z][:,d]  (K = S_), write [S,B,E] layout bf16
__global__ __launch_bounds__(256) void k_av(const __bf16* __restrict__ pr,
                                            const __bf16* __restrict__ vt,
                                            unsigned short* __restrict__ o) {
    const int z = blockIdx.z;
    const int wid = threadIdx.x >> 5;          // 8 waves stacked along M
    const int m0 = blockIdx.x * 512 + wid * 64;
    v8f acc[4][4]; zero_acc(acc);
    wmma_block(pr + ((size_t)z * S_ + m0) * S_,
               vt + (size_t)z * HD_ * S_, S_, S_, S_, acc);
    const int lane = threadIdx.x & 31, col = lane & 15, rb = (lane >> 4) * 8;
    const int b = z / H_, h = z % H_;
    #pragma unroll
    for (int i = 0; i < 4; ++i)
        #pragma unroll
        for (int j = 0; j < 4; ++j)
            #pragma unroll
            for (int v = 0; v < 8; ++v) {
                const int s = m0 + i * 16 + rb + v;
                const int d = j * 16 + col;    // 0..63
                o[((size_t)s * B_ + b) * E_ + h * HD_ + d] = f2bf(acc[i][j][v]);
            }
}

// out-proj + bias + residual(x) -> f32
__global__ __launch_bounds__(256) void k_outproj(const __bf16* __restrict__ A,
                                                 const __bf16* __restrict__ W,
                                                 const float* __restrict__ bias,
                                                 const float* __restrict__ xres,
                                                 float* __restrict__ y) {
    __shared__ __bf16 sA[2 * 128 * 32];
    const int wid = threadIdx.x >> 5, wm = wid >> 2, wn = wid & 3;
    const int mB = blockIdx.x * 128;
    const int m0 = mB + wm * 64;
    const int n0 = blockIdx.y * 256 + wn * 64;
    v8f acc[4][4]; zero_acc(acc);
    wmma_block_async(A + (size_t)mB * E_, wm * 64, W + (size_t)n0 * E_,
                     E_, E_, E_, sA, acc);
    const int lane = threadIdx.x & 31, col = lane & 15, rb = (lane >> 4) * 8;
    #pragma unroll
    for (int i = 0; i < 4; ++i)
        #pragma unroll
        for (int j = 0; j < 4; ++j)
            #pragma unroll
            for (int v = 0; v < 8; ++v) {
                const int t = m0 + i * 16 + rb + v;
                const int n = n0 + j * 16 + col;
                y[(size_t)t * E_ + n] = acc[i][j][v] + bias[n] + xres[(size_t)t * E_ + n];
            }
}

// LayerNorm(src) -> out f32 (+ bf16 copy)
__global__ __launch_bounds__(256) void k_ln(const float* __restrict__ src,
                                            const float* __restrict__ g,
                                            const float* __restrict__ b,
                                            float* __restrict__ out,
                                            unsigned short* __restrict__ outbf) {
    __shared__ float sm[16];
    const size_t base = (size_t)blockIdx.x * E_;
    float v[4], s = 0.f, s2 = 0.f;
    #pragma unroll
    for (int i = 0; i < 4; ++i) {
        v[i] = src[base + threadIdx.x + i * 256];
        s += v[i]; s2 += v[i] * v[i];
    }
    s  = bredux(s, sm, false);
    s2 = bredux(s2, sm, false);
    const float mu  = s * (1.0f / E_);
    const float var = s2 * (1.0f / E_) - mu * mu;
    const float inv = rsqrtf(var + 1e-5f);
    #pragma unroll
    for (int i = 0; i < 4; ++i) {
        const int n = threadIdx.x + i * 256;
        const float y = (v[i] - mu) * inv * g[n] + b[n];
        out[base + n] = y;
        outbf[base + n] = f2bf(y);
    }
}

// gate logits + argmax (top-1, first-max tiebreak like jnp.argmax)
__global__ __launch_bounds__(256) void k_gate(const float* __restrict__ x1,
                                              const float* __restrict__ gw,
                                              const float* __restrict__ gb,
                                              int* __restrict__ eid) {
    __shared__ float lg[NE_];
    const int t = blockIdx.x;
    const int wid = threadIdx.x >> 5, lane = threadIdx.x & 31;
    const float* xr = x1 + (size_t)t * E_;
    const float* wr = gw + (size_t)wid * E_;
    float p = 0.f;
    for (int k = lane; k < E_; k += 32) p += xr[k] * wr[k];
    #pragma unroll
    for (int m = 16; m; m >>= 1) p += __shfl_xor(p, m, 32);
    if (lane == 0) lg[wid] = p + gb[wid];
    __syncthreads();
    if (threadIdx.x == 0) {
        int best = 0; float bv = lg[0];
        #pragma unroll
        for (int e = 1; e < NE_; ++e)
            if (lg[e] > bv) { bv = lg[e]; best = e; }
        eid[t] = best;
    }
}

// routing control: ctl[0..7]=counts, ctl[8..15]=offsets, ctl[16..23]=cursors
__global__ void k_zero(int* __restrict__ ctl) {
    if (threadIdx.x < 32) ctl[threadIdx.x] = 0;
}
__global__ __launch_bounds__(256) void k_hist(const int* __restrict__ eid,
                                              int* __restrict__ ctl) {
    const int t = blockIdx.x * blockDim.x + threadIdx.x;
    if (t < T_) atomicAdd(&ctl[eid[t]], 1);
}
__global__ void k_scan(int* __restrict__ ctl) {
    if (threadIdx.x == 0) {
        int acc = 0;
        for (int e = 0; e < NE_; ++e) {
            const int c = ctl[e];
            ctl[8 + e] = acc;
            ctl[16 + e] = acc;
            acc += c;
        }
    }
}
// gather token rows into expert-sorted Xg; record row_of[token]
__global__ __launch_bounds__(256) void k_gather(const unsigned short* __restrict__ x1bf,
                                                const int* __restrict__ eid,
                                                int* __restrict__ ctl,
                                                int* __restrict__ row_of,
                                                unsigned short* __restrict__ Xg) {
    __shared__ int sp;
    const int t = blockIdx.x;
    if (threadIdx.x == 0) {
        sp = atomicAdd(&ctl[16 + eid[t]], 1);
        row_of[t] = sp;
    }
    __syncthreads();
    const int pos = sp;
    for (int i = threadIdx.x; i < E_; i += 256)
        Xg[(size_t)pos * E_ + i] = x1bf[(size_t)t * E_ + i];
}

// grouped FFN1: Hg = gelu(Xg @ w1[e]^T + b1[e])   (K = E_)
__global__ __launch_bounds__(256) void k_ffn1(const __bf16* __restrict__ Xg,
                                              const __bf16* __restrict__ w1,
                                              const float* __restrict__ b1,
                                              const int* __restrict__ ctl,
                                              unsigned short* __restrict__ Hg) {
    __shared__ __bf16 sA[2 * 128 * 32];
    const int e = blockIdx.z;
    const int cnt = ctl[e], off = ctl[8 + e];
    const int m0 = blockIdx.x * 128;
    if (m0 >= cnt) return;
    const int wid = threadIdx.x >> 5, wm = wid >> 2, wn = wid & 3;
    const int mW = m0 + wm * 64;                       // segment-relative
    const int n0 = blockIdx.y * 256 + wn * 64;
    v8f acc[4][4]; zero_acc(acc);
    wmma_block_async(Xg + (size_t)(off + m0) * E_, wm * 64,
                     w1 + ((size_t)e * DFF_ + n0) * E_, E_, E_, E_, sA, acc);
    const int lane = threadIdx.x & 31, col = lane & 15, rb = (lane >> 4) * 8;
    #pragma unroll
    for (int i = 0; i < 4; ++i)
        #pragma unroll
        for (int j = 0; j < 4; ++j)
            #pragma unroll
            for (int v = 0; v < 8; ++v) {
                const int r = mW + i * 16 + rb + v;
                if (r < cnt) {
                    const int n = n0 + j * 16 + col;
                    const float val = gelu_exact(acc[i][j][v] + b1[e * DFF_ + n]);
                    Hg[(size_t)(off + r) * DFF_ + n] = f2bf(val);
                }
            }
}

// grouped FFN2: Yg = Hg @ w2[e]^T + b2[e]   (K = DFF_), f32 out
__global__ __launch_bounds__(256) void k_ffn2(const __bf16* __restrict__ Hg,
                                              const __bf16* __restrict__ w2,
                                              const float* __restrict__ b2,
                                              const int* __restrict__ ctl,
                                              float* __restrict__ Yg) {
    __shared__ __bf16 sA[2 * 128 * 32];
    const int e = blockIdx.z;
    const int cnt = ctl[e], off = ctl[8 + e];
    const int m0 = blockIdx.x * 128;
    if (m0 >= cnt) return;
    const int wid = threadIdx.x >> 5, wm = wid >> 2, wn = wid & 3;
    const int mW = m0 + wm * 64;
    const int n0 = blockIdx.y * 256 + wn * 64;
    v8f acc[4][4]; zero_acc(acc);
    wmma_block_async(Hg + (size_t)(off + m0) * DFF_, wm * 64,
                     w2 + ((size_t)e * E_ + n0) * DFF_, DFF_, DFF_, DFF_, sA, acc);
    const int lane = threadIdx.x & 31, col = lane & 15, rb = (lane >> 4) * 8;
    #pragma unroll
    for (int i = 0; i < 4; ++i)
        #pragma unroll
        for (int j = 0; j < 4; ++j)
            #pragma unroll
            for (int v = 0; v < 8; ++v) {
                const int r = mW + i * 16 + rb + v;
                if (r < cnt) {
                    const int n = n0 + j * 16 + col;
                    Yg[(size_t)(off + r) * E_ + n] = acc[i][j][v] + b2[e * E_ + n];
                }
            }
}

// out = LayerNorm(x1 + Yg[row_of[t]])
__global__ __launch_bounds__(256) void k_final(const float* __restrict__ x1,
                                               const float* __restrict__ Yg,
                                               const int* __restrict__ row_of,
                                               const float* __restrict__ g,
                                               const float* __restrict__ b,
                                               float* __restrict__ out) {
    __shared__ float sm[16];
    const int t = blockIdx.x;
    const size_t base = (size_t)t * E_;
    const size_t rb_ = (size_t)row_of[t] * E_;
    float v[4], s = 0.f, s2 = 0.f;
    #pragma unroll
    for (int i = 0; i < 4; ++i) {
        const int n = threadIdx.x + i * 256;
        v[i] = x1[base + n] + Yg[rb_ + n];
        s += v[i]; s2 += v[i] * v[i];
    }
    s  = bredux(s, sm, false);
    s2 = bredux(s2, sm, false);
    const float mu  = s * (1.0f / E_);
    const float var = s2 * (1.0f / E_) - mu * mu;
    const float inv = rsqrtf(var + 1e-5f);
    #pragma unroll
    for (int i = 0; i < 4; ++i) {
        const int n = threadIdx.x + i * 256;
        out[base + n] = (v[i] - mu) * inv * g[n] + b[n];
    }
}

// ===================== launch =====================
extern "C" void kernel_launch(void* const* d_in, const int* in_sizes, int n_in,
                              void* d_out, int out_size, void* d_ws, size_t ws_size,
                              hipStream_t stream) {
    (void)in_sizes; (void)n_in; (void)out_size; (void)ws_size;
    const float* x    = (const float*)d_in[0];
    const float* wqkv = (const float*)d_in[1];
    const float* bqkv = (const float*)d_in[2];
    const float* wout = (const float*)d_in[3];
    const float* bout = (const float*)d_in[4];
    const float* ln1g = (const float*)d_in[5];
    const float* ln1b = (const float*)d_in[6];
    const float* ln2g = (const float*)d_in[7];
    const float* ln2b = (const float*)d_in[8];
    const float* gw   = (const float*)d_in[9];
    const float* gb   = (const float*)d_in[10];
    const float* w1   = (const float*)d_in[11];
    const float* b1   = (const float*)d_in[12];
    const float* w2   = (const float*)d_in[13];
    const float* b2   = (const float*)d_in[14];
    float* out = (float*)d_out;

    char* base = (char*)d_ws;
    size_t off = 0;
    auto take = [&](size_t bytes) -> void* {
        void* p = base + off;
        off += (bytes + 255) & ~(size_t)255;
        return p;
    };
    void* xbf    = take((size_t)T_ * E_ * 2);
    void* wqkvb  = take((size_t)3 * E_ * E_ * 2);
    void* woutb  = take((size_t)E_ * E_ * 2);
    void* w1b    = take((size_t)NE_ * DFF_ * E_ * 2);
    void* w2b    = take((size_t)NE_ * E_ * DFF_ * 2);
    void* qb     = take((size_t)T_ * E_ * 2);
    void* kb     = take((size_t)T_ * E_ * 2);
    void* vtb    = take((size_t)T_ * E_ * 2);
    void* probs  = take((size_t)B_ * H_ * S_ * S_ * 2);
    void* attnb  = take((size_t)T_ * E_ * 2);
    void* resid1 = take((size_t)T_ * E_ * 4);
    void* x1     = take((size_t)T_ * E_ * 4);
    void* x1b    = take((size_t)T_ * E_ * 2);
    void* eid    = take((size_t)T_ * 4);
    void* rowof  = take((size_t)T_ * 4);
    void* ctl    = take(32 * 4);
    void* Xg     = take((size_t)(T_ + PADR) * E_ * 2);
    void* Hg     = take((size_t)(T_ + PADR) * DFF_ * 2);
    void* Yg     = take((size_t)(T_ + PADR) * E_ * 4);

    // 1. fp32 -> bf16 conversions
    k_cvt<<<2048, 256, 0, stream>>>(x,    (unsigned short*)xbf,   T_ * E_);
    k_cvt<<<2048, 256, 0, stream>>>(wqkv, (unsigned short*)wqkvb, 3 * E_ * E_);
    k_cvt<<<2048, 256, 0, stream>>>(wout, (unsigned short*)woutb, E_ * E_);
    k_cvt<<<4096, 256, 0, stream>>>(w1,   (unsigned short*)w1b,   NE_ * DFF_ * E_);
    k_cvt<<<4096, 256, 0, stream>>>(w2,   (unsigned short*)w2b,   NE_ * E_ * DFF_);

    // 2. attention
    k_qkv<<<dim3(T_ / 128, (3 * E_) / 256), 256, 0, stream>>>(
        (const __bf16*)xbf, (const __bf16*)wqkvb, bqkv,
        (unsigned short*)qb, (unsigned short*)kb, (unsigned short*)vtb);
    k_scores<<<dim3(S_ / 128, S_ / 256, B_ * H_), 256, 0, stream>>>(
        (const __bf16*)qb, (const __bf16*)kb, (unsigned short*)probs);
    k_softmax<<<B_ * H_ * S_, 256, 0, stream>>>((unsigned short*)probs);
    k_av<<<dim3(S_ / 512, 1, B_ * H_), 256, 0, stream>>>(
        (const __bf16*)probs, (const __bf16*)vtb, (unsigned short*)attnb);
    k_outproj<<<dim3(T_ / 128, E_ / 256), 256, 0, stream>>>(
        (const __bf16*)attnb, (const __bf16*)woutb, bout, x, (float*)resid1);
    k_ln<<<T_, 256, 0, stream>>>((const float*)resid1, ln1g, ln1b,
                                 (float*)x1, (unsigned short*)x1b);

    // 3. MoE: route, gather, grouped FFN, combine
    k_gate<<<T_, 256, 0, stream>>>((const float*)x1, gw, gb, (int*)eid);
    k_zero<<<1, 32, 0, stream>>>((int*)ctl);
    k_hist<<<T_ / 256, 256, 0, stream>>>((const int*)eid, (int*)ctl);
    k_scan<<<1, 1, 0, stream>>>((int*)ctl);
    k_gather<<<T_, 256, 0, stream>>>((const unsigned short*)x1b, (const int*)eid,
                                     (int*)ctl, (int*)rowof, (unsigned short*)Xg);
    k_ffn1<<<dim3(T_ / 128, DFF_ / 256, NE_), 256, 0, stream>>>(
        (const __bf16*)Xg, (const __bf16*)w1b, b1, (const int*)ctl,
        (unsigned short*)Hg);
    k_ffn2<<<dim3(T_ / 128, E_ / 256, NE_), 256, 0, stream>>>(
        (const __bf16*)Hg, (const __bf16*)w2b, b2, (const int*)ctl, (float*)Yg);
    k_final<<<T_, 256, 0, stream>>>((const float*)x1, (const float*)Yg,
                                    (const int*)rowof, ln2g, ln2b, out);
}
